// MultiHeadSelfAttention_38216619000199
// MI455X (gfx1250) — matrix operand order repaired and checked by reading, not compiled, not run on hardware
//
#include <hip/hip_runtime.h>
#include <hip/hip_bf16.h>

// ---------------- problem constants ----------------
#define BB 2
#define TT 2048
#define DD 1024
#define HH 16
#define DHH 64
#define N3 3072
// ln(10000)/32
#define LOG1E4_OVER_32 0.28782313665f

typedef __attribute__((ext_vector_type(16))) __bf16 v16bf;
typedef __attribute__((ext_vector_type(8)))  float  v8f;
typedef __attribute__((ext_vector_type(4)))  unsigned u32x4;
typedef __attribute__((ext_vector_type(8)))  unsigned u32x8;

struct FragBF {
    union {
        v16bf v;
        uint4 q[2];
        unsigned u[8];
        unsigned short us[16];
    };
};

static __device__ __forceinline__ unsigned short f2bf(float f) {
    union { float f; unsigned u; } c; c.f = f;
    unsigned r = c.u + 0x7fffu + ((c.u >> 16) & 1u);   // round-to-nearest-even
    return (unsigned short)(r >> 16);
}

#if defined(__has_builtin)
#if __has_builtin(__builtin_amdgcn_cvt_pk_bf16_f32)
#define HAVE_PK_BF16 1
#endif
#endif

// pack two f32 -> 2x bf16 in one dword
static __device__ __forceinline__ unsigned pkbf(float a, float b) {
#ifdef HAVE_PK_BF16
    auto r = __builtin_amdgcn_cvt_pk_bf16_f32(a, b);
    unsigned u; __builtin_memcpy(&u, &r, 4); return u;
#else
    return (unsigned)f2bf(a) | ((unsigned)f2bf(b) << 16);
#endif
}

static __device__ __forceinline__ v8f wmma_bf16(v16bf a, v16bf b, v8f c) {
    return __builtin_amdgcn_wmma_f32_16x16x32_bf16(
        false, a, false, b, (short)0, c, false, false);
}

// A fragment (16x32 bf16, M x K), src row-major [row][k], ld in elements.
static __device__ __forceinline__ FragBF load_a_bf(const unsigned short* p, int ld, int lane) {
    FragBF f;
    int m = lane & 15, h = lane >> 4;
    const unsigned short* b = p + (size_t)m * ld + h * 8;
    f.q[0] = *(const uint4*)(b);
    f.q[1] = *(const uint4*)(b + 16);
    return f;
}

// B fragment (32x16 bf16, K x N), source stored n-major: element (k,n) at p[n*ld+k].
static __device__ __forceinline__ FragBF load_b_nmajor(const unsigned short* p, int ld, int lane) {
    FragBF f;
    int n = lane & 15, h = lane >> 4;
    const unsigned short* b = p + (size_t)n * ld + h * 8;
    f.q[0] = *(const uint4*)(b);
    f.q[1] = *(const uint4*)(b + 16);
    return f;
}

// Tensor Data Mover: 2-D bf16 tile load, global -> LDS.
// D# group0/group1 per CDNA5 ISA.
static __device__ __forceinline__ void tdm_load_2d_bf16(
    unsigned lds_addr, unsigned long long gaddr,
    unsigned dim0, unsigned dim1, unsigned stride0)
{
    u32x4 g0;
    g0[0] = 0x1u;                                              // count=1 valid descriptor
    g0[1] = lds_addr;                                          // lds_addr
    g0[2] = (unsigned)(gaddr & 0xffffffffull);                 // global_addr[31:0]
    g0[3] = (unsigned)((gaddr >> 32) & 0x01ffffffull) | (2u << 30); // addr[56:32] | type=2
    u32x8 g1;
    g1[0] = (1u << 16);                                        // data_size=2B, wg_mask=0
    g1[1] = (dim0 & 0xffffu) << 16;                            // tensor_dim0 lo
    g1[2] = ((dim0 >> 16) & 0xffffu) | ((dim1 & 0xffffu) << 16); // dim0 hi | tensor_dim1 lo
    g1[3] = ((dim1 >> 16) & 0xffffu) | ((dim0 & 0xffffu) << 16); // dim1 hi | tile_dim0
    g1[4] = (dim1 & 0xffffu);                                  // tile_dim1 (tile_dim2=0)
    g1[5] = stride0;                                           // tensor_dim0_stride lo32
    g1[6] = 0;
    g1[7] = 0;
    asm volatile("tensor_load_to_lds %0, %1" :: "s"(g0), "s"(g1) : "memory");
}

// =====================================================================
// Kernel 1: QKV projection (x @ qkv_w) + RoPE, output q/k/v bf16 [b,h,t,dh]
// =====================================================================
__global__ __launch_bounds__(256) void qkv_rope_kernel(
    const float* __restrict__ x, const float* __restrict__ w,
    unsigned short* __restrict__ qb, unsigned short* __restrict__ kb,
    unsigned short* __restrict__ vb)
{
    const int wave = threadIdx.x >> 5;
    const int lane = threadIdx.x & 31;
    const int wid  = blockIdx.x * 8 + wave;
    const int strip = wid % 48;
    const int rtile = wid / 48;
    const int n0 = strip * 64;
    const int bt0 = rtile * 16;
    const int mlane = lane & 15, half = lane >> 4;

    __shared__ __align__(16) unsigned short sw[8][64 * 32];
    unsigned short* wT = sw[wave];

    v8f acc[4] = {{}, {}, {}, {}};

    for (int k0 = 0; k0 < DD; k0 += 32) {
        const float* wr = w + (size_t)(k0 + lane) * N3 + n0;
        __builtin_prefetch(wr + 32 * N3, 0, 0);
        #pragma unroll
        for (int nn = 0; nn < 64; nn += 4) {
            float4 f = *(const float4*)(wr + nn);
            wT[(nn + 0) * 32 + lane] = f2bf(f.x);
            wT[(nn + 1) * 32 + lane] = f2bf(f.y);
            wT[(nn + 2) * 32 + lane] = f2bf(f.z);
            wT[(nn + 3) * 32 + lane] = f2bf(f.w);
        }
        asm volatile("s_wait_dscnt 0x0" ::: "memory");

        FragBF a;
        const float* xr = x + (size_t)(bt0 + mlane) * DD + k0 + half * 8;
        float4 f0 = *(const float4*)(xr);
        float4 f1 = *(const float4*)(xr + 4);
        float4 f2 = *(const float4*)(xr + 16);
        float4 f3 = *(const float4*)(xr + 20);
        a.u[0] = pkbf(f0.x, f0.y); a.u[1] = pkbf(f0.z, f0.w);
        a.u[2] = pkbf(f1.x, f1.y); a.u[3] = pkbf(f1.z, f1.w);
        a.u[4] = pkbf(f2.x, f2.y); a.u[5] = pkbf(f2.z, f2.w);
        a.u[6] = pkbf(f3.x, f3.y); a.u[7] = pkbf(f3.z, f3.w);

        #pragma unroll
        for (int nt = 0; nt < 4; ++nt) {
            FragBF b = load_b_nmajor(wT + nt * 16 * 32, 32, lane);
            acc[nt] = wmma_bf16(a.v, b.v, acc[nt]);
        }
        asm volatile("s_wait_dscnt 0x0" ::: "memory");
    }

    const int sec = strip / 16;   // 0=q 1=k 2=v
    const int hh  = strip % 16;
    unsigned short* dst = (sec == 0) ? qb : ((sec == 1) ? kb : vb);

    #pragma unroll
    for (int ct = 0; ct < 4; ++ct) {
        int dh = ct * 16 + mlane;
        float invf = __expf(-(float)(dh & 31) * LOG1E4_OVER_32);
        #pragma unroll
        for (int r = 0; r < 8; ++r) {
            int bt = bt0 + r + 8 * half;
            int b  = bt >> 11;
            int t  = bt & 2047;
            float val = acc[ct][r];
            if (sec < 2) {
                float ang = (float)t * invf;
                float cv = __cosf(ang), sv = __sinf(ang);
                float other = acc[ct ^ 2][r];
                float rot = (ct < 2) ? -other : other;
                val = val * cv + rot * sv;
            }
            dst[((size_t)(b * HH + hh) * TT + t) * DHH + dh] = f2bf(val);
        }
    }
}

// =====================================================================
// Kernel 2: causal flash attention. One wave = 16 q rows of one (b,h).
// K tile: async engine global->LDS (ASYNCcnt). V tile: TDM descriptor
// load (TENSORcnt) issued early, waited only after S+softmax.
// 4 waves / block => 36KB LDS.
// =====================================================================
__global__ __launch_bounds__(128) void flash_attn_kernel(
    const unsigned short* __restrict__ qg, const unsigned short* __restrict__ kg,
    const unsigned short* __restrict__ vg, unsigned short* __restrict__ ctx)
{
    const int wave = threadIdx.x >> 5;
    const int lane = threadIdx.x & 31;
    const int wid  = blockIdx.x * 4 + wave;
    const int bh = wid >> 7;
    const int qt = wid & 127;
    const int t0 = qt * 16;
    const int b  = bh >> 4, h = bh & 15;
    const int mlane = lane & 15, half = lane >> 4;

    __shared__ __align__(16) unsigned short skr[4][32 * 64]; // K tile [key][d]
    __shared__ __align__(16) unsigned short svr[4][32 * 64]; // V tile [key][d]
    __shared__ __align__(16) unsigned short spt[4][16 * 32]; // P [m][k]
    unsigned short* Kr = skr[wave];
    unsigned short* Vr = svr[wave];
    unsigned short* Pt = spt[wave];
    const unsigned klds = (unsigned)(uintptr_t)(void*)Kr + (unsigned)lane * 16u;
    const unsigned vlds = (unsigned)(uintptr_t)(void*)Vr;   // uniform: TDM descriptor

    const unsigned short* qp = qg + (size_t)bh * TT * DHH;
    const unsigned short* kp = kg + (size_t)bh * TT * DHH;
    const unsigned short* vp = vg + (size_t)bh * TT * DHH;

    FragBF aq[2];
    #pragma unroll
    for (int c = 0; c < 2; ++c)
        aq[c] = load_a_bf(qp + (size_t)t0 * DHH + c * 32, DHH, lane);

    float mrow[8], lrow[8];
    v8f oacc[4] = {{}, {}, {}, {}};
    #pragma unroll
    for (int r = 0; r < 8; ++r) { mrow[r] = -3.0e38f; lrow[r] = 0.0f; }

    for (int kbk = 0; kbk < t0 + 16; kbk += 32) {
        // ---- issue V tile via Tensor Data Mover (waited late) ----
        tdm_load_2d_bf16(vlds,
                         (unsigned long long)(uintptr_t)(vp + (size_t)kbk * DHH),
                         64u, 32u, 64u);
        // ---- issue K tile via async engine (8 x 512B) ----
        {
            unsigned long long g =
                (unsigned long long)(uintptr_t)(kp + (size_t)kbk * DHH) + (unsigned long long)lane * 16ull;
            #pragma unroll
            for (int i = 0; i < 8; ++i) {
                asm volatile("global_load_async_to_lds_b128 %0, %1, off"
                             :: "v"(klds + i * 512u), "v"(g + (unsigned long long)(i * 512))
                             : "memory");
            }
        }
        asm volatile("s_wait_asynccnt 0x0" ::: "memory");
        // ---- S = Q K^T for 32 keys (two 16-col tiles) from LDS ----
        v8f s[2];
        #pragma unroll
        for (int tile = 0; tile < 2; ++tile) {
            v8f st = {};
            #pragma unroll
            for (int c = 0; c < 2; ++c) {
                FragBF bk = load_b_nmajor(Kr + tile * 16 * DHH + c * 32, DHH, lane);
                st = wmma_bf16(aq[c].v, bk.v, st);
            }
            s[tile] = st;
        }
        // scale + causal mask
        #pragma unroll
        for (int tile = 0; tile < 2; ++tile) {
            int col = kbk + tile * 16 + mlane;
            #pragma unroll
            for (int r = 0; r < 8; ++r) {
                int row = t0 + r + 8 * half;
                float v = s[tile][r] * 0.125f;
                s[tile][r] = (col > row) ? -3.0e38f : v;
            }
        }
        // ---- online softmax over the 32 columns ----
        #pragma unroll
        for (int r = 0; r < 8; ++r) {
            float mx = fmaxf(s[0][r], s[1][r]);
            #pragma unroll
            for (int off = 1; off < 16; off <<= 1)
                mx = fmaxf(mx, __shfl_xor(mx, off, 32));
            float mnew = fmaxf(mrow[r], mx);
            float corr = __expf(mrow[r] - mnew);
            float p0 = __expf(s[0][r] - mnew);
            float p1 = __expf(s[1][r] - mnew);
            s[0][r] = p0; s[1][r] = p1;
            float sum = p0 + p1;
            #pragma unroll
            for (int off = 1; off < 16; off <<= 1)
                sum += __shfl_xor(sum, off, 32);
            lrow[r] = lrow[r] * corr + sum;
            mrow[r] = mnew;
            #pragma unroll
            for (int ct = 0; ct < 4; ++ct) oacc[ct][r] *= corr;
        }
        // ---- P -> LDS (A layout) ----
        #pragma unroll
        for (int tile = 0; tile < 2; ++tile)
            #pragma unroll
            for (int r = 0; r < 8; ++r)
                Pt[(r + 8 * half) * 32 + tile * 16 + mlane] = f2bf(s[tile][r]);
        asm volatile("s_wait_dscnt 0x0" ::: "memory");
        FragBF ap = load_a_bf(Pt, 32, lane);
        // ---- V tile ready? sync TDM, gather B fragments, O += P V ----
        __builtin_amdgcn_s_wait_tensorcnt(0);
        #pragma unroll
        for (int ct = 0; ct < 4; ++ct) {
            FragBF bv;
            #pragma unroll
            for (int i = 0; i < 16; ++i)
                bv.us[i] = Vr[((i >> 3) * 16 + half * 8 + (i & 7)) * DHH + ct * 16 + mlane];
            oacc[ct] = wmma_bf16(ap.v, bv.v, oacc[ct]);
        }
    }

    #pragma unroll
    for (int ct = 0; ct < 4; ++ct)
        #pragma unroll
        for (int r = 0; r < 8; ++r) {
            int bt = b * TT + t0 + r + 8 * half;
            float val = oacc[ct][r] / lrow[r];
            ctx[(size_t)bt * DD + h * DHH + ct * 16 + mlane] = f2bf(val);
        }
}

// =====================================================================
// Kernel 3: output projection ctx @ out_w + out_b  -> f32 [B,T,D]
// =====================================================================
__global__ __launch_bounds__(256) void out_proj_kernel(
    const unsigned short* __restrict__ ctx, const float* __restrict__ w,
    const float* __restrict__ bias, float* __restrict__ out)
{
    const int wave = threadIdx.x >> 5;
    const int lane = threadIdx.x & 31;
    const int wid  = blockIdx.x * 8 + wave;
    const int strip = wid & 15;
    const int rtile = wid >> 4;
    const int n0 = strip * 64;
    const int bt0 = rtile * 16;
    const int mlane = lane & 15, half = lane >> 4;

    __shared__ __align__(16) unsigned short sw[8][64 * 32];
    unsigned short* wT = sw[wave];

    v8f acc[4] = {{}, {}, {}, {}};

    for (int k0 = 0; k0 < DD; k0 += 32) {
        const float* wr = w + (size_t)(k0 + lane) * DD + n0;
        __builtin_prefetch(wr + 32 * DD, 0, 0);
        #pragma unroll
        for (int nn = 0; nn < 64; nn += 4) {
            float4 f = *(const float4*)(wr + nn);
            wT[(nn + 0) * 32 + lane] = f2bf(f.x);
            wT[(nn + 1) * 32 + lane] = f2bf(f.y);
            wT[(nn + 2) * 32 + lane] = f2bf(f.z);
            wT[(nn + 3) * 32 + lane] = f2bf(f.w);
        }
        asm volatile("s_wait_dscnt 0x0" ::: "memory");

        FragBF a = load_a_bf(ctx + (size_t)bt0 * DD + k0, DD, lane);
        #pragma unroll
        for (int nt = 0; nt < 4; ++nt) {
            FragBF bfr = load_b_nmajor(wT + nt * 16 * 32, 32, lane);
            acc[nt] = wmma_bf16(a.v, bfr.v, acc[nt]);
        }
        asm volatile("s_wait_dscnt 0x0" ::: "memory");
    }

    #pragma unroll
    for (int ct = 0; ct < 4; ++ct) {
        int col = n0 + ct * 16 + mlane;
        float bv = bias[col];
        #pragma unroll
        for (int r = 0; r < 8; ++r) {
            int bt = bt0 + r + 8 * half;
            out[(size_t)bt * DD + col] = acc[ct][r] + bv;
        }
    }
}

// =====================================================================
extern "C" void kernel_launch(void* const* d_in, const int* in_sizes, int n_in,
                              void* d_out, int out_size, void* d_ws, size_t ws_size,
                              hipStream_t stream) {
    const float* x      = (const float*)d_in[0];
    // d_in[1] = mask (bool causal) -- handled analytically
    const float* qkv_w  = (const float*)d_in[2];
    const float* out_w  = (const float*)d_in[3];
    const float* out_b  = (const float*)d_in[4];
    float* out = (float*)d_out;

    const size_t SQ = (size_t)BB * HH * TT * DHH;   // 4,194,304 elems
    unsigned short* ws = (unsigned short*)d_ws;
    unsigned short* qb  = ws;
    unsigned short* kb  = ws + SQ;
    unsigned short* vb  = ws + 2 * SQ;
    unsigned short* ctx = ws + 3 * SQ;

    dim3 blk256(256), blk128(128);
    qkv_rope_kernel<<<1536, blk256, 0, stream>>>(x, qkv_w, qb, kb, vb);
    flash_attn_kernel<<<1024, blk128, 0, stream>>>(qb, kb, vb, ctx);
    out_proj_kernel<<<512, blk256, 0, stream>>>(ctx, out_w, out_b, out);
}
